// IDB_87093346828411
// MI455X (gfx1250) — compile-verified, matrix operand order checked by reading, not run on hardware
//
#include <hip/hip_runtime.h>
#include <hip/hip_bf16.h>

typedef __attribute__((ext_vector_type(16))) __bf16        v16bf;
typedef __attribute__((ext_vector_type(8)))  float         v8f;
typedef __attribute__((ext_vector_type(8)))  unsigned int  v8u;
typedef __attribute__((ext_vector_type(4)))  unsigned int  v4u;
typedef __attribute__((ext_vector_type(8)))  int           v8i;
typedef __attribute__((ext_vector_type(4)))  int           v4i;

#define WMMA_BF16(a, b, c) \
    __builtin_amdgcn_wmma_f32_16x16x32_bf16(false, (a), false, (b), (short)0, (c), false, false)

#if __has_builtin(__builtin_amdgcn_tensor_load_to_lds) && __has_builtin(__builtin_amdgcn_s_wait_tensorcnt)
#define HAVE_TDM 1
#else
#define HAVE_TDM 0
#endif

// Pair index (u32 word w of a v16bf fragment holds bf16 elements for
// K = 2*pair, 2*pair+1) per CDNA5 ISA 7.12.2 16-bit A/B layouts.
__device__ __forceinline__ int pair_of(int lane, int w) {
    return ((w & 4) << 1) | (w & 3) | ((lane & 16) >> 2);
}

// Split two fp32 values into pair-packed bf16 hi and residual-lo words.
__device__ __forceinline__ void split2(float v0, float v1, unsigned& hi, unsigned& lo) {
    __bf16 h0 = (__bf16)v0, h1 = (__bf16)v1;
    __bf16 l0 = (__bf16)(v0 - (float)h0), l1 = (__bf16)(v1 - (float)h1);
    hi = (unsigned)__builtin_bit_cast(unsigned short, h0) |
         ((unsigned)__builtin_bit_cast(unsigned short, h1) << 16);
    lo = (unsigned)__builtin_bit_cast(unsigned short, l0) |
         ((unsigned)__builtin_bit_cast(unsigned short, l1) << 16);
}

#if HAVE_TDM
// Issue one TDM 2D tile load (tile_h rows of tile_w fp32 elements, row stride
// in elements) into LDS at byte offset lds_off. D# layout per CDNA5 ISA ch.8.
__device__ __forceinline__ void tdm_load_2d(unsigned lds_off, const void* gptr,
                                            unsigned tile_w, unsigned tile_h,
                                            unsigned long long row_stride,
                                            unsigned tensor_w, unsigned tensor_h) {
    unsigned long long ga = (unsigned long long)(size_t)gptr;
    v4u g0 = { 1u,                                   // count=1 (valid user D#)
               lds_off,                              // lds_addr
               (unsigned)(ga & 0xFFFFFFFFu),         // global_addr[31:0]
               (unsigned)((ga >> 32) & 0x01FFFFFFull) | 0x80000000u }; // [56:32] | type=2
    v8i g1 = { (int)0x00020000u,                     // data_size = 4 bytes
               (int)((tensor_w & 0xFFFFu) << 16),    // tensor_dim0[15:0]
               (int)((tensor_w >> 16) | ((tensor_h & 0xFFFFu) << 16)),
               (int)(((tensor_h >> 16) & 0xFFFFu) | (tile_w << 16)),  // tile_dim0
               (int)tile_h,                          // tile_dim1 (tile_dim2 = 0)
               (int)(unsigned)row_stride,            // tensor_dim0_stride[31:0]
               (int)((row_stride >> 32) & 0xFFFFull),
               0 };
    v4i gz = { 0, 0, 0, 0 };
#if defined(__clang_major__) && __clang_major__ >= 23
    v8i gz8 = { 0, 0, 0, 0, 0, 0, 0, 0 };
    __builtin_amdgcn_tensor_load_to_lds(g0, g1, gz, gz, gz8, 0);
#else
    __builtin_amdgcn_tensor_load_to_lds(g0, g1, gz, gz, 0);
#endif
}
#endif

// ---------------------------------------------------------------------------
// Kernel 1: 3x3 conv (64->64, pad 1) + ReLU, implicit GEMM via bf16 WMMA.
// Block = 128 threads = 4 waves; tile = 64 output pixels of one row.
// M = 16 pixels/wave, N = 64 channels (4 tiles), K = 576 (18 chunks of 32).
// Weight K-slices streamed into LDS by the Tensor Data Mover.
// ---------------------------------------------------------------------------
__global__ __launch_bounds__(128) void conv3x3_relu_wmma(
    const float* __restrict__ x, const float* __restrict__ w1,
    const float* __restrict__ b1, float* __restrict__ y)
{
    __shared__ unsigned b_raw[64 * 32];      // raw f32 weight slice (TDM dest)
    __shared__ unsigned b_hi[64 * 16];       // [n][pair] packed bf16 hi
    __shared__ unsigned b_lo[64 * 16];       // [n][pair] packed bf16 lo
    __shared__ unsigned a_hi[64 * 16];       // [m][pair]
    __shared__ unsigned a_lo[64 * 16];

    const int tid  = threadIdx.x;
    const int lane = tid & 31;
    const int wave = tid >> 5;

    const int blk = blockIdx.x;              // 4 * 512 * 8 = 16384 blocks
    const int b   = blk >> 12;
    const int rem = blk & 4095;
    const int oh  = rem >> 3;
    const int ow0 = (rem & 7) << 6;

    const float* xb = x + ((size_t)b << 24); // b*64*512*512
    const int m_tile = (wave << 4) + (lane & 15);
    const int nlo    = lane & 15;

    v8f acc[4];
    #pragma unroll
    for (int t = 0; t < 4; ++t)
        #pragma unroll
        for (int r = 0; r < 8; ++r) acc[t][r] = 0.0f;

    for (int kc = 0; kc < 18; ++kc) {
        const int k0 = kc << 5;
        __syncthreads();   // previous chunk's LDS consumers are done

#if HAVE_TDM
        if (wave == 0) {
            tdm_load_2d((unsigned)(size_t)(void*)b_raw, w1 + k0,
                        /*tile_w=*/32, /*tile_h=*/64, /*row_stride=*/576,
                        /*tensor_w=*/576, /*tensor_h=*/64);
            __builtin_amdgcn_s_wait_tensorcnt(0);
        }
#else
        for (int i = tid; i < 64 * 32; i += 128)
            b_raw[i] = __builtin_bit_cast(unsigned, w1[(i >> 5) * 576 + k0 + (i & 31)]);
#endif

        // Stage A pairs straight from global (coalesced along m).
        for (int i = tid; i < 1024; i += 128) {
            int pk = i >> 6, m = i & 63;
            int k  = k0 + (pk << 1);
            int c0 = k / 9,        r0 = k - c0 * 9,  kh0 = r0 / 3, kw0 = r0 - kh0 * 3;
            int k1 = k + 1;
            int c1 = k1 / 9,       r1 = k1 - c1 * 9, kh1 = r1 / 3, kw1 = r1 - kh1 * 3;
            int row0 = oh - 1 + kh0, col0 = ow0 - 1 + m + kw0;
            int row1 = oh - 1 + kh1, col1 = ow0 - 1 + m + kw1;
            float v0 = 0.0f, v1 = 0.0f;
            if ((unsigned)row0 < 512u && (unsigned)col0 < 512u)
                v0 = xb[((size_t)c0 << 18) + (row0 << 9) + col0];
            if ((unsigned)row1 < 512u && (unsigned)col1 < 512u)
                v1 = xb[((size_t)c1 << 18) + (row1 << 9) + col1];
            unsigned h, l; split2(v0, v1, h, l);
            a_hi[m * 16 + pk] = h;
            a_lo[m * 16 + pk] = l;
        }
        __syncthreads();   // TDM data + A pairs visible

        // Convert raw weight slice into packed bf16 hi/lo pairs.
        for (int i = tid; i < 1024; i += 128) {
            int n = i >> 4, pk = i & 15;
            float v0 = __builtin_bit_cast(float, b_raw[(n << 5) + (pk << 1)]);
            float v1 = __builtin_bit_cast(float, b_raw[(n << 5) + (pk << 1) + 1]);
            unsigned h, l; split2(v0, v1, h, l);
            b_hi[(n << 4) + pk] = h;
            b_lo[(n << 4) + pk] = l;
        }
        __syncthreads();

        // Fragments: pure ds_load_b32 of pre-split pairs.
        v8u awh, awl;
        #pragma unroll
        for (int w = 0; w < 8; ++w) {
            int pk = pair_of(lane, w);
            awh[w] = a_hi[(m_tile << 4) + pk];
            awl[w] = a_lo[(m_tile << 4) + pk];
        }
        v16bf Ah = __builtin_bit_cast(v16bf, awh);
        v16bf Al = __builtin_bit_cast(v16bf, awl);

        #pragma unroll
        for (int t = 0; t < 4; ++t) {
            int n = (t << 4) + nlo;
            v8u bwh, bwl;
            #pragma unroll
            for (int w = 0; w < 8; ++w) {
                int pk = pair_of(lane, w);
                bwh[w] = b_hi[(n << 4) + pk];
                bwl[w] = b_lo[(n << 4) + pk];
            }
            v16bf Bh = __builtin_bit_cast(v16bf, bwh);
            v16bf Bl = __builtin_bit_cast(v16bf, bwl);
            acc[t] = WMMA_BF16(Ah, Bh, acc[t]);
            acc[t] = WMMA_BF16(Ah, Bl, acc[t]);
            acc[t] = WMMA_BF16(Al, Bh, acc[t]);
        }
    }

    // D layout: vgpr r -> M = r (lanes 0-15) / r+8 (lanes 16-31); N = lane&15
    const int lhalf = lane >> 4;
    float* yb = y + ((size_t)b << 24);
    #pragma unroll
    for (int t = 0; t < 4; ++t) {
        int ch = (t << 4) + nlo;
        float bias = b1[ch];
        #pragma unroll
        for (int r = 0; r < 8; ++r) {
            int m  = r + (lhalf << 3);
            int ow = ow0 + (wave << 4) + m;
            yb[((size_t)ch << 18) + (oh << 9) + ow] = fmaxf(acc[t][r] + bias, 0.0f);
        }
    }
}

// ---------------------------------------------------------------------------
// Kernel 2: avgpool2x2 + 1x1 reduce (64->16) + BN(inference) + ReLU
// ---------------------------------------------------------------------------
__global__ __launch_bounds__(256) void reduce_bn_relu(
    const float* __restrict__ y, const float* __restrict__ wr,
    const float* __restrict__ br, const float* __restrict__ gamma,
    const float* __restrict__ beta, const float* __restrict__ mean,
    const float* __restrict__ var, float* __restrict__ t)
{
    int idx = blockIdx.x * 256 + threadIdx.x;     // 4*256*256 = 262144
    int b  = idx >> 16;
    int hw = idx & 65535;
    int h  = hw >> 8;
    int w  = hw & 255;

    const float* yb = y + ((size_t)b << 24);
    float xp[64];
    #pragma unroll 4
    for (int c = 0; c < 64; ++c) {
        size_t base = ((size_t)c << 18) + ((size_t)(h << 1) << 9) + (w << 1);
        xp[c] = 0.25f * (yb[base] + yb[base + 1] + yb[base + 512] + yb[base + 513]);
    }
    for (int r = 0; r < 16; ++r) {
        float s = br[r];
        #pragma unroll 8
        for (int c = 0; c < 64; ++c) s += wr[(r << 6) + c] * xp[c];
        s = (s - mean[r]) * rsqrtf(var[r] + 1e-5f) * gamma[r] + beta[r];
        t[(((size_t)b * 16 + r) << 16) + hw] = fmaxf(s, 0.0f);
    }
}

// ---------------------------------------------------------------------------
// Kernel 3: 1x1 span conv (16 -> 36) producing involution weights.
// ---------------------------------------------------------------------------
__global__ __launch_bounds__(256) void span_conv(
    const float* __restrict__ t, const float* __restrict__ wspan,
    const float* __restrict__ bs, float* __restrict__ wgt)
{
    int idx = blockIdx.x * 256 + threadIdx.x;     // 262144
    int b  = idx >> 16;
    int hw = idx & 65535;

    float tt[16];
    #pragma unroll
    for (int r = 0; r < 16; ++r)
        tt[r] = t[(((size_t)b * 16 + r) << 16) + hw];

    for (int s = 0; s < 36; ++s) {
        float a = bs[s];
        #pragma unroll
        for (int r = 0; r < 16; ++r) a += wspan[(s << 4) + r] * tt[r];
        wgt[(((size_t)b * 36 + s) << 16) + hw] = a;
    }
}

// ---------------------------------------------------------------------------
// Kernel 4: involution (K=3, stride=2, G=4, GC=16) + ReLU -> z
// ---------------------------------------------------------------------------
__global__ __launch_bounds__(256) void involution_relu(
    const float* __restrict__ y, const float* __restrict__ wgt,
    float* __restrict__ z)
{
    int idx = blockIdx.x * 256 + threadIdx.x;     // 4*4*256*256 = 1048576
    int b   = idx >> 18;
    int rem = idx & 262143;
    int g   = rem >> 16;
    int hw  = rem & 65535;
    int h   = hw >> 8;
    int w   = hw & 255;

    float wk[9];
    #pragma unroll
    for (int kk = 0; kk < 9; ++kk)
        wk[kk] = wgt[(((size_t)b * 36 + g * 9 + kk) << 16) + hw];

    for (int gc = 0; gc < 16; ++gc) {
        int ch = (g << 4) + gc;
        const float* yc = y + (((size_t)b * 64 + ch) << 18);
        float s = 0.0f;
        #pragma unroll
        for (int kh = 0; kh < 3; ++kh) {
            int ih = (h << 1) - 1 + kh;
            if ((unsigned)ih >= 512u) continue;
            #pragma unroll
            for (int kw = 0; kw < 3; ++kw) {
                int iw = (w << 1) - 1 + kw;
                if ((unsigned)iw >= 512u) continue;
                s += wk[kh * 3 + kw] * yc[(ih << 9) + iw];
            }
        }
        z[(((size_t)b * 64 + ch) << 16) + hw] = fmaxf(s, 0.0f);
    }
}

// ---------------------------------------------------------------------------
// Kernel 5: final 1x1 conv (64->64) + ReLU via bf16 WMMA GEMM.
// Block = 128 threads = 4 waves; tile = 64 consecutive pixels (same b, h).
// z tile streamed into LDS by the Tensor Data Mover.
// ---------------------------------------------------------------------------
__global__ __launch_bounds__(128) void conv1x1_relu_wmma(
    const float* __restrict__ z, const float* __restrict__ w3,
    const float* __restrict__ b3, float* __restrict__ out)
{
    __shared__ unsigned z_raw[64 * 64];      // raw f32 z tile (TDM dest)
    __shared__ unsigned z_hi[64 * 32];       // [m][pair]
    __shared__ unsigned z_lo[64 * 32];
    __shared__ unsigned w_hi[64 * 32];       // [n][pair]
    __shared__ unsigned w_lo[64 * 32];

    const int tid  = threadIdx.x;
    const int lane = tid & 31;
    const int wave = tid >> 5;

    const int blk = blockIdx.x;              // 4 * 1024 = 4096 blocks
    const int b   = blk >> 10;
    const int hw0 = (blk & 1023) << 6;

    const float* zb = z + ((size_t)b << 22); // b*64*65536

#if HAVE_TDM
    if (wave == 0) {
        tdm_load_2d((unsigned)(size_t)(void*)z_raw, zb + hw0,
                    /*tile_w=*/64, /*tile_h=*/64, /*row_stride=*/65536,
                    /*tensor_w=*/65536, /*tensor_h=*/64);
        __builtin_amdgcn_s_wait_tensorcnt(0);
    }
#else
    for (int i = tid; i < 4096; i += 128)
        z_raw[i] = __builtin_bit_cast(unsigned, zb[((size_t)(i >> 6) << 16) + hw0 + (i & 63)]);
#endif

    // Stage weight pairs from global (contiguous along K).
    for (int i = tid; i < 2048; i += 128) {
        int n = i >> 5, pk = i & 31;
        float v0 = w3[(n << 6) + (pk << 1)];
        float v1 = w3[(n << 6) + (pk << 1) + 1];
        unsigned h, l; split2(v0, v1, h, l);
        w_hi[(n << 5) + pk] = h;
        w_lo[(n << 5) + pk] = l;
    }
    __syncthreads();

    // Convert z tile: pairs along channel (K) dimension.
    for (int i = tid; i < 2048; i += 128) {
        int pk = i >> 6, m = i & 63;
        float v0 = __builtin_bit_cast(float, z_raw[((pk << 1) << 6) + m]);
        float v1 = __builtin_bit_cast(float, z_raw[((pk << 1) + 1 << 6) + m]);
        unsigned h, l; split2(v0, v1, h, l);
        z_hi[(m << 5) + pk] = h;
        z_lo[(m << 5) + pk] = l;
    }
    __syncthreads();

    const int m_tile = (wave << 4) + (lane & 15);
    const int nlo    = lane & 15;

    v8f acc[4];
    #pragma unroll
    for (int t = 0; t < 4; ++t)
        #pragma unroll
        for (int r = 0; r < 8; ++r) acc[t][r] = 0.0f;

    #pragma unroll
    for (int kc = 0; kc < 2; ++kc) {
        v8u awh, awl;
        #pragma unroll
        for (int w = 0; w < 8; ++w) {
            int pk = (kc << 4) + pair_of(lane, w);
            awh[w] = z_hi[(m_tile << 5) + pk];
            awl[w] = z_lo[(m_tile << 5) + pk];
        }
        v16bf Ah = __builtin_bit_cast(v16bf, awh);
        v16bf Al = __builtin_bit_cast(v16bf, awl);
        #pragma unroll
        for (int t = 0; t < 4; ++t) {
            int n = (t << 4) + nlo;
            v8u bwh, bwl;
            #pragma unroll
            for (int w = 0; w < 8; ++w) {
                int pk = (kc << 4) + pair_of(lane, w);
                bwh[w] = w_hi[(n << 5) + pk];
                bwl[w] = w_lo[(n << 5) + pk];
            }
            v16bf Bh = __builtin_bit_cast(v16bf, bwh);
            v16bf Bl = __builtin_bit_cast(v16bf, bwl);
            acc[t] = WMMA_BF16(Ah, Bh, acc[t]);
            acc[t] = WMMA_BF16(Ah, Bl, acc[t]);
            acc[t] = WMMA_BF16(Al, Bh, acc[t]);
        }
    }

    const int lhalf = lane >> 4;
    float* ob = out + ((size_t)b << 22);
    #pragma unroll
    for (int t = 0; t < 4; ++t) {
        int ch = (t << 4) + nlo;
        float bias = b3[ch];
        #pragma unroll
        for (int r = 0; r < 8; ++r) {
            int m = r + (lhalf << 3);
            ob[((size_t)ch << 16) + hw0 + m] = fmaxf(acc[t][r] + bias, 0.0f);
        }
    }
}

// ---------------------------------------------------------------------------
extern "C" void kernel_launch(void* const* d_in, const int* in_sizes, int n_in,
                              void* d_out, int out_size, void* d_ws, size_t ws_size,
                              hipStream_t stream) {
    const float* x     = (const float*)d_in[0];
    const float* w1    = (const float*)d_in[1];
    const float* b1    = (const float*)d_in[2];
    const float* wr    = (const float*)d_in[3];
    const float* br    = (const float*)d_in[4];
    const float* gamma = (const float*)d_in[5];
    const float* beta  = (const float*)d_in[6];
    const float* mean  = (const float*)d_in[7];
    const float* var   = (const float*)d_in[8];
    const float* wspan = (const float*)d_in[9];
    const float* bs    = (const float*)d_in[10];
    const float* w3    = (const float*)d_in[11];
    const float* b3    = (const float*)d_in[12];
    float* out = (float*)d_out;

    float* wsf = (float*)d_ws;
    float* y   = wsf;                      // 4*64*512*512 = 67108864 floats
    float* t   = y   + 67108864;           // 4*16*256*256 =  4194304 floats
    float* wgt = t   + 4194304;            // 4*36*256*256 =  9437184 floats
    float* zz  = wgt + 9437184;            // 4*64*256*256 = 16777216 floats

    conv3x3_relu_wmma<<<16384, 128, 0, stream>>>(x, w1, b1, y);
    reduce_bn_relu   <<<1024, 256, 0, stream>>>(y, wr, br, gamma, beta, mean, var, t);
    span_conv        <<<1024, 256, 0, stream>>>(t, wspan, bs, wgt);
    involution_relu  <<<4096, 256, 0, stream>>>(y, wgt, zz);
    conv1x1_relu_wmma<<<4096, 128, 0, stream>>>(zz, w3, b3, out);
}
